// Self_Attn_3813930959393
// MI455X (gfx1250) — matrix operand order repaired and checked by reading, not compiled, not run on hardware
//
#include <hip/hip_runtime.h>

// ---------------------------------------------------------------------------
// Types for CDNA5 WMMA
// ---------------------------------------------------------------------------
typedef __attribute__((ext_vector_type(16))) __bf16 v16bf;
typedef __attribute__((ext_vector_type(8)))  float  v8f;
typedef __attribute__((ext_vector_type(4)))  int    v4i;   // true vector (not HIP struct)

union Frag {
    v16bf v;
    __bf16 h[16];
    uint4 q[2];
};

__device__ __forceinline__ v8f wmma_bf16(const Frag& a, const Frag& b, v8f c) {
    return __builtin_amdgcn_wmma_f32_16x16x32_bf16(
        /*neg_a=*/false, a.v, /*neg_b=*/false, b.v,
        /*c_mod=*/(short)0, c, /*reuse_a=*/false, /*reuse_b=*/false);
}

#define EPS_BN 1e-5f

// ---------------------------------------------------------------------------
// Optional gfx1250 async global->LDS staging (guarded; falls back to b128 copy)
// ---------------------------------------------------------------------------
#if defined(__has_builtin)
#if __has_builtin(__builtin_amdgcn_global_load_async_to_lds_b128)
#define USE_ASYNC_LDS 1
#endif
#endif

#ifdef USE_ASYNC_LDS
typedef __attribute__((address_space(1))) v4i gv4i;
typedef __attribute__((address_space(3))) v4i lv4i;
#endif

__device__ __forceinline__ void lds_copy16(__bf16* l, const __bf16* g) {
#ifdef USE_ASYNC_LDS
    __builtin_amdgcn_global_load_async_to_lds_b128(
        (gv4i*)(unsigned long long)(const void*)g,
        (lv4i*)(unsigned int)(unsigned long long)(void*)l,
        0, 0);
#else
    *(uint4*)l = *(const uint4*)g;
#endif
}

__device__ __forceinline__ void lds_copy_wait() {
#ifdef USE_ASYNC_LDS
    asm volatile("s_wait_asynccnt 0" ::: "memory");
#endif
}

// ---------------------------------------------------------------------------
// Kernel 1: weight prep (bf16 conversion, natural [O][C] layout, qkv fused).
// wc1 [256][256], wqkv [320][256] (rows: 32 q, 32 k, 256 v), wc2 [256][256].
// ---------------------------------------------------------------------------
__global__ __launch_bounds__(256) void prep_kernel(
    const float* __restrict__ c1w,
    const float* __restrict__ qw, const float* __restrict__ qb,
    const float* __restrict__ kw, const float* __restrict__ kb2,
    const float* __restrict__ vw, const float* __restrict__ vb,
    const float* __restrict__ c2w,
    __bf16* __restrict__ wc1,
    __bf16* __restrict__ wqkv,
    __bf16* __restrict__ wc2,
    float* __restrict__ bqkv)
{
    int idx = blockIdx.x * blockDim.x + threadIdx.x;   // < 320*256
    if (idx < 320 * 256) {
        int o = idx >> 8, c = idx & 255;
        float w = (o < 32) ? qw[o * 256 + c]
                : (o < 64) ? kw[(o - 32) * 256 + c]
                           : vw[(o - 64) * 256 + c];
        wqkv[idx] = (__bf16)w;
    }
    if (idx < 256 * 256) {
        wc1[idx] = (__bf16)c1w[idx];
        wc2[idx] = (__bf16)c2w[idx];
    }
    if (idx < 320) {
        bqkv[idx] = (idx < 32) ? qb[idx] : (idx < 64) ? kb2[idx - 32] : vb[idx - 64];
    }
}

// ---------------------------------------------------------------------------
// Kernel 2: bilinear downsample 128x128 -> 64x64 (align_corners=True).
// x [B,C,128,128] f32 -> down [B, pos=4096, C] bf16.
// ---------------------------------------------------------------------------
__global__ __launch_bounds__(64) void downsample_kernel(
    const float* __restrict__ x, __bf16* __restrict__ down)
{
    int ox = threadIdx.x;      // 0..63
    int oy = blockIdx.x;       // 0..63
    int c  = blockIdx.y;       // 0..255
    int b  = blockIdx.z;       // 0..3
    float fy = oy * (127.0f / 63.0f);
    float fx = ox * (127.0f / 63.0f);
    int y0 = (int)floorf(fy), x0 = (int)floorf(fx);
    int y1 = min(y0 + 1, 127), x1 = min(x0 + 1, 127);
    float wy = fy - (float)y0, wx = fx - (float)x0;
    const float* xb = x + ((size_t)b * 256 + c) * 128 * 128;
    float r0 = xb[y0 * 128 + x0] * (1.0f - wx) + xb[y0 * 128 + x1] * wx;
    float r1 = xb[y1 * 128 + x0] * (1.0f - wx) + xb[y1 * 128 + x1] * wx;
    float v  = r0 * (1.0f - wy) + r1 * wy;
    down[((size_t)b * 4096 + oy * 64 + ox) * 256 + c] = (__bf16)v;
}

// ---------------------------------------------------------------------------
// Kernel 3: tiled WMMA GEMM  out[b] = A[b] (MxK bf16) x W^T, W given [N][K] bf16.
// Block tile 128x64, K-step 32, 8 waves as 4(M)x2(N), each wave 32x32 (2x2 WMMA).
// All LDS fragment reads are aligned b128.
// Epilogue (templated):
//   MODE 0: BN+ReLU -> f32 outF + bf16 outB           (conv1)
//   MODE 1: +bias; cols<64 -> qk bf16 [pos][64],
//           cols>=64 -> vT bf16 [ch][pos]             (qkv)
//   MODE 2: BN -> f32 outF                            (conv2)
// ---------------------------------------------------------------------------
template <int MODE>
__global__ __launch_bounds__(256) void gemm_kernel(
    const __bf16* __restrict__ A, const __bf16* __restrict__ Wt,
    int M, int N, int K, long aBatchStride,
    float* __restrict__ outF, __bf16* __restrict__ outB,
    __bf16* __restrict__ outB2,
    const float* __restrict__ scale, const float* __restrict__ bias)
{
    __shared__ __align__(16) __bf16 Alds[128 * 32];
    __shared__ __align__(16) __bf16 Blds[64 * 32];   // [n][k]

    const int b      = blockIdx.z;
    const int mBlock = blockIdx.x * 128;
    const int nBlock = blockIdx.y * 64;
    const __bf16* Ab = A + (size_t)b * aBatchStride;

    const int tid  = threadIdx.x;
    const int lane = tid & 31;
    const int wid  = tid >> 5;
    const int wm   = (wid & 3) * 32;
    const int wn   = (wid >> 2) * 32;
    const int hlf  = lane >> 4;      // 0/1
    const int l15  = lane & 15;

    // staging roles
    const int arow = tid >> 1, acs = (tid & 1) * 16;   // A: 128 rows x 32 k
    const int brow = tid >> 2, bcs = (tid & 3) * 8;    // B: 64 rows x 32 k

    v8f zero = {0.f, 0.f, 0.f, 0.f, 0.f, 0.f, 0.f, 0.f};
    v8f acc[2][2] = {{zero, zero}, {zero, zero}};

    const int ks  = hlf * 8;    // A-operand K-group interleave (ISA layout)
    const int kb0 = hlf * 8;    // B-operand K-group interleave (same as A)

    for (int kb = 0; kb < K; kb += 32) {
        const __bf16* aSrc = Ab + (size_t)(mBlock + arow) * K + kb + acs;
        const __bf16* bSrc = Wt + (size_t)(nBlock + brow) * K + kb + bcs;
        lds_copy16(Alds + arow * 32 + acs,      aSrc);
        lds_copy16(Alds + arow * 32 + acs + 8,  aSrc + 8);
        lds_copy16(Blds + brow * 32 + bcs,      bSrc);
        if (kb + 32 < K) {   // prefetch next K tiles into cache
            __builtin_prefetch(aSrc + 32, 0, 1);
            __builtin_prefetch(bSrc + 32, 0, 1);
        }
        lds_copy_wait();
        __syncthreads();

        Frag af[2], bfr[2];
        #pragma unroll
        for (int mi = 0; mi < 2; mi++) {
            int r = wm + mi * 16 + l15;
            af[mi].q[0] = *(const uint4*)(Alds + r * 32 + ks);
            af[mi].q[1] = *(const uint4*)(Alds + r * 32 + 16 + ks);
        }
        #pragma unroll
        for (int ni = 0; ni < 2; ni++) {
            int col = wn + ni * 16 + l15;
            bfr[ni].q[0] = *(const uint4*)(Blds + col * 32 + kb0);
            bfr[ni].q[1] = *(const uint4*)(Blds + col * 32 + kb0 + 16);
        }
        acc[0][0] = wmma_bf16(af[0], bfr[0], acc[0][0]);
        acc[0][1] = wmma_bf16(af[0], bfr[1], acc[0][1]);
        acc[1][0] = wmma_bf16(af[1], bfr[0], acc[1][0]);
        acc[1][1] = wmma_bf16(af[1], bfr[1], acc[1][1]);
        __syncthreads();
    }

    const float invs = rsqrtf(1.0f + EPS_BN);
    #pragma unroll
    for (int mi = 0; mi < 2; mi++) {
        #pragma unroll
        for (int ni = 0; ni < 2; ni++) {
            int colg = nBlock + wn + ni * 16 + l15;
            float sc = (MODE != 1) ? scale[colg] * invs : 0.0f;
            float bi = bias[colg];
            #pragma unroll
            for (int r = 0; r < 8; r++) {
                int rowg = mBlock + wm + mi * 16 + r + hlf * 8;
                float v = acc[mi][ni][r];
                if (MODE == 0) {
                    v = fmaxf(v * sc + bi, 0.0f);
                    size_t oi = (size_t)b * M * N + (size_t)rowg * N + colg;
                    outF[oi] = v;
                    outB[oi] = (__bf16)v;
                } else if (MODE == 1) {
                    v += bi;
                    if (colg < 64) {
                        outB[(size_t)b * M * 64 + (size_t)rowg * 64 + colg] = (__bf16)v;
                    } else {
                        outB2[(size_t)b * 256 * M + (size_t)(colg - 64) * M + rowg] = (__bf16)v;
                    }
                } else {
                    size_t oi = (size_t)b * M * N + (size_t)rowg * N + colg;
                    outF[oi] = v * sc + bi;
                }
            }
        }
    }
}

// ---------------------------------------------------------------------------
// Kernel 4: fused flash-attention + residual + BN2 + ReLU.
// qk [B,4096,64] bf16 (q:0..31, k:32..63), vT [B,256,4096] bf16.
// Workgroup = 4 independent waves sharing one 16-row query tile; each wave owns
// 64 output channels. No block barriers in the main loop.
// Row sums of P are computed with an extra WMMA against a ones matrix (replaces
// 32 ds_bpermute ops per chunk and normalizes over the bf16-rounded P, exactly
// matching the PV numerator).
// y2 = relu(bn2(gamma * softmax(q k^T) v + y1))  -> bf16 [B,4096,256]
// ---------------------------------------------------------------------------
__global__ __launch_bounds__(128) void flash_attn_kernel(
    const __bf16* __restrict__ qk,
    const __bf16* __restrict__ vT,
    const float* __restrict__ y1,
    __bf16* __restrict__ y2,
    const float* __restrict__ gamma_p,
    const float* __restrict__ bn2_s, const float* __restrict__ bn2_b)
{
    __shared__ __align__(16) __bf16 Plds[4][16 * 32];

    const int b    = blockIdx.y;
    const int i0   = blockIdx.x * 16;
    const int tid  = threadIdx.x;
    const int lane = tid & 31;
    const int wid  = tid >> 5;
    const int hlf  = lane >> 4;
    const int l15  = lane & 15;
    const int chBase = wid * 64;
    const size_t qkb = (size_t)b * 4096 * 64;
    const size_t vb  = (size_t)b * 256 * 4096;
    const int ks  = hlf * 8;
    const int kb0 = hlf * 8;

    // Q fragment (16 rows x 32 d), loaded once.
    Frag qf;
    {
        const __bf16* qp = qk + qkb + (size_t)(i0 + l15) * 64;
        qf.q[0] = *(const uint4*)(qp + ks);
        qf.q[1] = *(const uint4*)(qp + 16 + ks);
    }

    // all-ones B fragment: P x ones -> row sums broadcast across D columns
    Frag ones;
    #pragma unroll
    for (int t = 0; t < 16; t++) ones.h[t] = (__bf16)1.0f;

    v8f zero = {0.f, 0.f, 0.f, 0.f, 0.f, 0.f, 0.f, 0.f};
    v8f O[4] = {zero, zero, zero, zero};
    float mrow[8], lrow[8];
    #pragma unroll
    for (int r = 0; r < 8; r++) { mrow[r] = -3.0e38f; lrow[r] = 0.0f; }

    for (int j0 = 0; j0 < 4096; j0 += 32) {
        // S = q x k^T for 32 key columns (two 16x16 WMMA tiles)
        Frag kf0, kf1;
        {
            const __bf16* p0 = qk + qkb + (size_t)(j0 + l15) * 64 + 32 + kb0;
            const __bf16* p1 = qk + qkb + (size_t)(j0 + 16 + l15) * 64 + 32 + kb0;
            kf0.q[0] = *(const uint4*)(p0); kf0.q[1] = *(const uint4*)(p0 + 16);
            kf1.q[0] = *(const uint4*)(p1); kf1.q[1] = *(const uint4*)(p1 + 16);
            __builtin_prefetch(p0 + 32 * 64, 0, 1);   // next chunk's keys
        }
        v8f s0 = wmma_bf16(qf, kf0, zero);
        v8f s1 = wmma_bf16(qf, kf1, zero);

        // online softmax: running max via xor-shuffle reduce across 16-lane halves
        float scale8[8];
        #pragma unroll
        for (int r = 0; r < 8; r++) {
            float t = fmaxf(s0[r], s1[r]);
            t = fmaxf(t, __shfl_xor(t, 1));
            t = fmaxf(t, __shfl_xor(t, 2));
            t = fmaxf(t, __shfl_xor(t, 4));
            t = fmaxf(t, __shfl_xor(t, 8));
            float mn = fmaxf(mrow[r], t);
            float sc = __expf(mrow[r] - mn);
            mrow[r] = mn;
            s0[r] = __expf(s0[r] - mn);
            s1[r] = __expf(s1[r] - mn);
            scale8[r] = sc;
        }

        // relayout P: D-layout -> A-fragment via per-wave LDS (wave-local)
        __bf16* P = Plds[wid];
        #pragma unroll
        for (int r = 0; r < 8; r++) {
            int row = r + hlf * 8;
            P[row * 32 + l15]      = (__bf16)s0[r];
            P[row * 32 + 16 + l15] = (__bf16)s1[r];
        }
        asm volatile("s_wait_dscnt 0" ::: "memory");
        Frag pf;
        pf.q[0] = *(const uint4*)(P + l15 * 32 + ks);
        pf.q[1] = *(const uint4*)(P + l15 * 32 + 16 + ks);

        // row sums of (bf16-rounded) P via matrix unit; update normalizer
        v8f rs = wmma_bf16(pf, ones, zero);
        #pragma unroll
        for (int r = 0; r < 8; r++) lrow[r] = lrow[r] * scale8[r] + rs[r];

        // O = O*scale + P x V  (4 channel tiles per wave, V direct from vT)
        #pragma unroll
        for (int ct = 0; ct < 4; ct++) {
            #pragma unroll
            for (int r = 0; r < 8; r++) O[ct][r] *= scale8[r];
            Frag vf;
            const __bf16* vp = vT + vb + (size_t)(chBase + ct * 16 + l15) * 4096 + j0 + kb0;
            vf.q[0] = *(const uint4*)(vp);
            vf.q[1] = *(const uint4*)(vp + 16);
            O[ct] = wmma_bf16(pf, vf, O[ct]);
        }
    }

    // epilogue: normalize, gamma*out + y1, BN2, ReLU -> bf16
    const float gamma = *gamma_p;
    const float invs  = rsqrtf(1.0f + EPS_BN);
    const size_t ob   = (size_t)b * 4096 * 256;
    #pragma unroll
    for (int ct = 0; ct < 4; ct++) {
        int col = chBase + ct * 16 + l15;
        float se = bn2_s[col] * invs, be = bn2_b[col];
        #pragma unroll
        for (int r = 0; r < 8; r++) {
            int row = i0 + r + hlf * 8;
            float v = O[ct][r] / lrow[r];
            float o = gamma * v + y1[ob + (size_t)row * 256 + col];
            float a = fmaxf(o * se + be, 0.0f);
            y2[ob + (size_t)row * 256 + col] = (__bf16)a;
        }
    }
}

// ---------------------------------------------------------------------------
// Kernel 5: bilinear upsample 64x64 -> 128x128 (align_corners=True)
// + residual + ReLU.  y3 [B,4096,256] f32, x/out [B,C,128,128] f32.
// ---------------------------------------------------------------------------
__global__ __launch_bounds__(128) void upsample_kernel(
    const float* __restrict__ y3, const float* __restrict__ x,
    float* __restrict__ out)
{
    int ox = threadIdx.x;      // 0..127
    int oy = blockIdx.x;       // 0..127
    int c  = blockIdx.y;       // 0..255
    int b  = blockIdx.z;       // 0..3
    float fy = oy * (63.0f / 127.0f);
    float fx = ox * (63.0f / 127.0f);
    int y0 = (int)floorf(fy), x0 = (int)floorf(fx);
    int y1 = min(y0 + 1, 63), x1 = min(x0 + 1, 63);
    float wy = fy - (float)y0, wx = fx - (float)x0;
    const float* yb = y3 + (size_t)b * 4096 * 256;
    float g00 = yb[(size_t)(y0 * 64 + x0) * 256 + c];
    float g01 = yb[(size_t)(y0 * 64 + x1) * 256 + c];
    float g10 = yb[(size_t)(y1 * 64 + x0) * 256 + c];
    float g11 = yb[(size_t)(y1 * 64 + x1) * 256 + c];
    float r0 = g00 * (1.0f - wx) + g01 * wx;
    float r1 = g10 * (1.0f - wx) + g11 * wx;
    float v  = r0 * (1.0f - wy) + r1 * wy;
    size_t xi = (((size_t)b * 256 + c) * 128 + oy) * 128 + ox;
    out[xi] = fmaxf(v + x[xi], 0.0f);
}

// ---------------------------------------------------------------------------
// Host-side orchestration
// ---------------------------------------------------------------------------
extern "C" void kernel_launch(void* const* d_in, const int* in_sizes, int n_in,
                              void* d_out, int out_size, void* d_ws, size_t ws_size,
                              hipStream_t stream) {
    (void)in_sizes; (void)n_in; (void)out_size; (void)ws_size;
    const float* x       = (const float*)d_in[0];
    const float* conv1_w = (const float*)d_in[1];
    const float* bn1_s   = (const float*)d_in[2];
    const float* bn1_b   = (const float*)d_in[3];
    const float* q_w     = (const float*)d_in[4];
    const float* q_b     = (const float*)d_in[5];
    const float* k_w     = (const float*)d_in[6];
    const float* k_b     = (const float*)d_in[7];
    const float* v_w     = (const float*)d_in[8];
    const float* v_b     = (const float*)d_in[9];
    const float* gamma   = (const float*)d_in[10];
    const float* bn2_s   = (const float*)d_in[11];
    const float* bn2_b   = (const float*)d_in[12];
    const float* conv2_w = (const float*)d_in[13];
    const float* bn3_s   = (const float*)d_in[14];
    const float* bn3_b   = (const float*)d_in[15];
    float* out = (float*)d_out;

    const int B = 4, C = 256, NPOS = 4096, NQKV = 320;

    // carve workspace (256B-aligned slabs)
    char* ws = (char*)d_ws;
    size_t off = 0;
    auto carve = [&](size_t bytes) -> char* {
        char* p = ws + off;
        off = (off + bytes + 255) & ~(size_t)255;
        return p;
    };
    __bf16* down_bf = (__bf16*)carve((size_t)B * NPOS * C * 2);
    __bf16* wc1     = (__bf16*)carve((size_t)C * C * 2);
    __bf16* wqkv    = (__bf16*)carve((size_t)NQKV * C * 2);
    __bf16* wc2     = (__bf16*)carve((size_t)C * C * 2);
    float*  bqkv    = (float*)carve((size_t)NQKV * 4);
    float*  y1f     = (float*)carve((size_t)B * NPOS * C * 4);
    __bf16* y1bf    = (__bf16*)carve((size_t)B * NPOS * C * 2);
    __bf16* qkbf    = (__bf16*)carve((size_t)B * NPOS * 64 * 2);
    __bf16* vTbf    = (__bf16*)carve((size_t)B * C * NPOS * 2);
    __bf16* y2bf    = (__bf16*)carve((size_t)B * NPOS * C * 2);
    float*  y3f     = (float*)carve((size_t)B * NPOS * C * 4);

    // 1) weight prep (bf16, natural layout)
    prep_kernel<<<dim3((320 * 256 + 255) / 256), dim3(256), 0, stream>>>(
        conv1_w, q_w, q_b, k_w, k_b, v_w, v_b, conv2_w, wc1, wqkv, wc2, bqkv);

    // 2) downsample -> bf16 [B, 4096, 256]
    downsample_kernel<<<dim3(64, 256, B), dim3(64), 0, stream>>>(x, down_bf);

    // 3) conv1 + BN1 + ReLU : y1 = relu(bn1(down x W1^T))
    gemm_kernel<0><<<dim3(NPOS / 128, C / 64, B), dim3(256), 0, stream>>>(
        down_bf, wc1, NPOS, C, C, (long)NPOS * C,
        y1f, y1bf, nullptr, bn1_s, bn1_b);

    // 4) fused qkv : [q|k] -> qkbf [B,4096,64], v -> vTbf [B,256,4096]
    gemm_kernel<1><<<dim3(NPOS / 128, NQKV / 64, B), dim3(256), 0, stream>>>(
        y1bf, wqkv, NPOS, NQKV, C, (long)NPOS * C,
        nullptr, qkbf, vTbf, nullptr, bqkv);

    // 5) flash attention + residual + BN2 + ReLU -> y2 bf16
    flash_attn_kernel<<<dim3(NPOS / 16, B), dim3(128), 0, stream>>>(
        qkbf, vTbf, y1f, y2bf, gamma, bn2_s, bn2_b);

    // 6) conv2 + BN3 : y3 = bn3(y2 x W2^T)
    gemm_kernel<2><<<dim3(NPOS / 128, C / 64, B), dim3(256), 0, stream>>>(
        y2bf, wc2, NPOS, C, C, (long)NPOS * C,
        y3f, nullptr, nullptr, bn3_s, bn3_b);

    // 7) upsample + residual + ReLU -> out
    upsample_kernel<<<dim3(128, 256, B), dim3(128), 0, stream>>>(y3f, x, out);
}
